// LanzariniGeodeticEngine_68831145886568
// MI455X (gfx1250) — matrix operand order repaired and checked by reading, not compiled, not run on hardware
//
#include <hip/hip_runtime.h>
#include <hip/hip_bf16.h>

// ---------------------------------------------------------------------------
// Gaussian-kernel attention, single-pass flash style, CDNA5 (gfx1250) WMMA.
//   W    = exp(-(|q|^2+|k|^2-2 q.k))           in (0,1]
//   attn = softmax(W/8)  -> bounded arg => no running max needed
//   out  = attn @ V ; alpha = 0.1/(std(attn, ddof=1)+1e-6)
// One wave owns a 16-row Q tile; 4 waves/block share K/V LDS staging.
// ---------------------------------------------------------------------------

typedef __bf16  v16bf __attribute__((ext_vector_type(16)));
typedef __bf16  v8bf  __attribute__((ext_vector_type(8)));
typedef float   v8f   __attribute__((ext_vector_type(8)));

#define BATCH 4
#define SEQ   4096
#define DIMD  64
#define TILE_M 16
#define TILE_J 32
#define WAVES  4
#define NTILES (BATCH * SEQ / TILE_M)   // 1024 waves total

__device__ inline v16bf load2x8(const __bf16* p0, const __bf16* p1) {
    const v8bf a = *(const v8bf*)p0;
    const v8bf b = *(const v8bf*)p1;
    v16bf r;
#pragma unroll
    for (int i = 0; i < 8; ++i) { r[i] = a[i]; r[i + 8] = b[i]; }
    return r;
}

__global__ __launch_bounds__(WAVES * 32)
void gka_fused_kernel(const float* __restrict__ Q,
                      const float* __restrict__ K,
                      const float* __restrict__ V,
                      float* __restrict__ Out,
                      float* __restrict__ partials) {
    __shared__ __bf16 sK[TILE_J][DIMD];          // K rows j..j+31 (row major)
    __shared__ __bf16 sVT[DIMD][TILE_J];         // V tile transposed: sVT[d][r]
    __shared__ float  sk2[TILE_J];               // |k|^2 of staged rows
    __shared__ __bf16 sP[WAVES][TILE_M][TILE_J]; // per-wave P transpose buffer

    const int tid  = threadIdx.x;
    const int wave = tid >> 5;
    const int lane = tid & 31;
    const int hl   = lane >> 4;       // lane-half (0/1) per WMMA layouts
    const int l16  = lane & 15;

    const int tileId = blockIdx.x * WAVES + wave;   // 0..1023
    const int b  = tileId >> 8;                     // / (SEQ/TILE_M)
    const int q0 = (tileId & 255) << 4;             // tile's first Q row
    const float* Qb = Q + (size_t)b * SEQ * DIMD;
    const float* Kb = K + (size_t)b * SEQ * DIMD;
    const float* Vb = V + (size_t)b * SEQ * DIMD;

    // ---- Q A-fragments (two K=32 chunks covering D=64) + per-row |q|^2 ----
    v16bf aq[2];
    float q2p = 0.f;
    {
        const float* qrow = Qb + (size_t)(q0 + l16) * DIMD;
#pragma unroll
        for (int c = 0; c < 2; ++c) {
            const int kc = 32 * c;
#pragma unroll
            for (int i = 0; i < 8; ++i) {
                float x0 = qrow[kc + hl * 8 + i];
                float x1 = qrow[kc + 16 + hl * 8 + i];
                q2p += x0 * x0 + x1 * x1;
                aq[c][i]     = (__bf16)x0;
                aq[c][i + 8] = (__bf16)x1;
            }
        }
    }
    const float q2_lane = q2p + __shfl_xor(q2p, 16);   // full |q(row l16)|^2
    float q2r[8];
#pragma unroll
    for (int r = 0; r < 8; ++r) q2r[r] = __shfl(q2_lane, r + 8 * hl, 32);

    // ---- accumulators: O (16x64, C-layout) and per-row sum(p), sum(p^2) ----
    v8f oacc[4];
#pragma unroll
    for (int nt = 0; nt < 4; ++nt) { v8f z = {}; oacc[nt] = z; }
    float rs[8] = {};
    float rq[8] = {};

    for (int j = 0; j < SEQ; j += TILE_J) {
        __syncthreads();   // protect LDS reuse from previous iteration

        // -------- cooperative staging: K rows (+|k|^2) and transposed V ----
        if (tid < 32) {
            const float* krow = Kb + (size_t)(j + tid) * DIMD;
            float ss = 0.f;
#pragma unroll
            for (int c = 0; c < 8; ++c) {
                v8bf pk;
#pragma unroll
                for (int i = 0; i < 8; ++i) {
                    float x = krow[c * 8 + i];
                    ss += x * x;
                    pk[i] = (__bf16)x;
                }
                *(v8bf*)&sK[tid][c * 8] = pk;
            }
            sk2[tid] = ss;
        } else if (tid < 64) {
            const int r = tid - 32;
            const float* vrow = Vb + (size_t)(j + r) * DIMD;
#pragma unroll
            for (int d = 0; d < DIMD; ++d) sVT[d][r] = (__bf16)vrow[d];
        }
        __syncthreads();

        // -------- S = Q . K^T for two 16x16 tiles (K-dim = 64 = 2 WMMAs) ----
        v8f S[2];
#pragma unroll
        for (int t = 0; t < 2; ++t) {
            const __bf16* kr = &sK[16 * t + l16][0];
            v16bf b0 = load2x8(kr + hl * 8,      kr + 16 + hl * 8);
            v16bf b1 = load2x8(kr + 32 + hl * 8, kr + 48 + hl * 8);
            v8f c = {};
            c = __builtin_amdgcn_wmma_f32_16x16x32_bf16(false, aq[0], false, b0,
                                                        (short)0, c, false, false);
            c = __builtin_amdgcn_wmma_f32_16x16x32_bf16(false, aq[1], false, b1,
                                                        (short)0, c, false, false);
            S[t] = c;
        }

        // -------- elementwise: p = exp(exp(-dist^2)/8); stats; pack bf16 ----
#pragma unroll
        for (int t = 0; t < 2; ++t) {
            const float k2v = sk2[16 * t + l16];
#pragma unroll
            for (int r = 0; r < 8; ++r) {
                float dist = q2r[r] + k2v - 2.0f * S[t][r];
                dist = fmaxf(dist, 0.0f);
                float w = __expf(-dist);           // BETA = 1, MOD = 1
                float p = __expf(0.125f * w);      // 1/sqrt(64)
                rs[r] += p;
                rq[r] += p * p;
                // C-layout element (m = r + 8*hl, col = 16*t + l16)
                sP[wave][r + 8 * hl][16 * t + l16] = (__bf16)p;
            }
        }
        __syncthreads();   // P buffer visible in A-layout order

        // -------- O += P . V : one A-frag (16x32), four 16-wide d-tiles -----
        const __bf16* prow = &sP[wave][l16][0];
        v16bf ap = load2x8(prow + hl * 16, prow + hl * 16 + 8);
#pragma unroll
        for (int nt = 0; nt < 4; ++nt) {
            const __bf16* vt = &sVT[nt * 16 + l16][0] + hl * 16;
            v16bf bv = load2x8(vt, vt + 8);
            oacc[nt] = __builtin_amdgcn_wmma_f32_16x16x32_bf16(
                false, ap, false, bv, (short)0, oacc[nt], false, false);
        }
    }

    // ---- finish row sums across the 16-lane groups of the C layout --------
#pragma unroll
    for (int r = 0; r < 8; ++r) {
        float s = rs[r], q = rq[r];
        s += __shfl_xor(s, 8); q += __shfl_xor(q, 8);
        s += __shfl_xor(s, 4); q += __shfl_xor(q, 4);
        s += __shfl_xor(s, 2); q += __shfl_xor(q, 2);
        s += __shfl_xor(s, 1); q += __shfl_xor(q, 1);
        rs[r] = s; rq[r] = q;
    }

    // ---- normalize and store output (row m = r + 8*hl, col = nt*16+l16) ---
    float* outb = Out + (size_t)b * SEQ * DIMD;
#pragma unroll
    for (int nt = 0; nt < 4; ++nt) {
#pragma unroll
        for (int r = 0; r < 8; ++r) {
            const int m = r + 8 * hl;
            outb[(size_t)(q0 + m) * DIMD + nt * 16 + l16] = oacc[nt][r] / rs[r];
        }
    }

    // ---- deterministic partial of sum(attn^2): per row rq/rs^2 ------------
    float ps = 0.f;
#pragma unroll
    for (int r = 0; r < 8; ++r) ps += rq[r] / (rs[r] * rs[r]);
    const float wsum = ps + __shfl_xor(ps, 16);   // both lane-halves
    if (lane == 0) partials[tileId] = wsum;
}

// Fixed-order reduction of per-wave partials -> unbiased std -> alpha.
__global__ void gka_finalize_kernel(const float* __restrict__ partials,
                                    float* __restrict__ alpha_out) {
    if (threadIdx.x == 0 && blockIdx.x == 0) {
        double s2 = 0.0;
        for (int i = 0; i < NTILES; ++i) s2 += (double)partials[i];
        const double n   = (double)BATCH * SEQ * SEQ;   // 67,108,864
        const double sum = (double)BATCH * SEQ;         // softmax rows sum to 1
        double var = (s2 - sum * sum / n) / (n - 1.0);
        if (var < 0.0) var = 0.0;
        alpha_out[0] = (float)(0.1 / (sqrt(var) + 1e-6));
    }
}

extern "C" void kernel_launch(void* const* d_in, const int* in_sizes, int n_in,
                              void* d_out, int out_size, void* d_ws, size_t ws_size,
                              hipStream_t stream) {
    const float* Q = (const float*)d_in[0];
    const float* K = (const float*)d_in[1];
    const float* V = (const float*)d_in[2];
    float* out      = (float*)d_out;                       // [B,N,D] + alpha
    float* partials = (float*)d_ws;                        // 1024 floats
    float* alpha    = out + (size_t)BATCH * SEQ * DIMD;    // last element

    gka_fused_kernel<<<NTILES / WAVES, WAVES * 32, 0, stream>>>(Q, K, V, out, partials);
    gka_finalize_kernel<<<1, 32, 0, stream>>>(partials, alpha);
}
